// FourierAttention_22265110463188
// MI455X (gfx1250) — matrix operand order repaired and checked by reading, not compiled, not run on hardware
//
#include <hip/hip_runtime.h>
#include <hip/hip_bf16.h>

// ---------------------------------------------------------------------------
// FourierAttention on MI455X (gfx1250, wave32, WMMA)
// B=4, L=S=2048, H=8, E=64, Lf=1025 (padded to 1056 = 66*16 = 33*32)
// ---------------------------------------------------------------------------

#define BDIM   4
#define LDIM   2048
#define HDIM   8
#define EDIM   64
#define BH     (BDIM * HDIM)       // 32
#define LF     1025                // rfft bins
#define LFP    1056                // padded bins (33 * 32)
#define NSTEP  33                  // y-steps of 32 bins

typedef __attribute__((ext_vector_type(16))) _Float16 v16h;
typedef __attribute__((ext_vector_type(8)))  _Float16 v8h;
typedef __attribute__((ext_vector_type(8)))  float    v8f;

static __device__ __forceinline__ v8f wmma_f16(v16h a, v16h b, v8f c) {
  // D = A(16x32 f16) * B(32x16 f16) + C(16x16 f32)
  return __builtin_amdgcn_wmma_f32_16x16x32_f16(
      /*neg_a=*/false, a, /*neg_b=*/false, b,
      /*c_mod=*/(short)0, c, /*reuse_a=*/false, /*reuse_b=*/false);
}

// Build a v16h A-fragment from two 16-byte (8-half) global chunks.
static __device__ __forceinline__ v16h combine8(const _Float16* lo, const _Float16* hi) {
  v8h a = *(const v8h*)lo;
  v8h b = *(const v8h*)hi;
  v16h r;
#pragma unroll
  for (int i = 0; i < 8; ++i) { r[i] = a[i]; r[i + 8] = b[i]; }
  return r;
}

// ---------------------------------------------------------------------------
// 1) Tiled transpose: [B,L,H,E] -> timeT [bh][e][l]  (coalesced both sides)
// ---------------------------------------------------------------------------
__global__ __launch_bounds__(256)
void fa_transpose_in(const float* __restrict__ src, float* __restrict__ dst) {
  __shared__ float tile[32][33];
  const int bh = blockIdx.z, b = bh >> 3, h = bh & 7;
  const int l0 = blockIdx.x * 32, e0 = blockIdx.y * 32;
  const int tx = threadIdx.x, ty = threadIdx.y;
  const float* s = src + ((size_t)b * LDIM * HDIM + h) * EDIM;   // + l*H*E + e
#pragma unroll
  for (int k = 0; k < 4; ++k) {
    const int l = l0 + ty + 8 * k;
    tile[ty + 8 * k][tx] = s[(size_t)l * (HDIM * EDIM) + e0 + tx];
  }
  __syncthreads();
  float* d = dst + (size_t)bh * EDIM * LDIM;                     // [bh][e][l]
#pragma unroll
  for (int k = 0; k < 4; ++k) {
    const int e = e0 + ty + 8 * k;
    d[(size_t)e * LDIM + l0 + tx] = tile[tx][ty + 8 * k];
  }
}

// ---------------------------------------------------------------------------
// 2) Radix-2 DIT FFT, 2048 points, one block (128 thr) per (bh,e) row.
//    Real input -> bins 0..1024, ortho scale 1/sqrt(2048). Pads zeroed.
// ---------------------------------------------------------------------------
static __device__ __forceinline__ unsigned brev11(unsigned v) {
  return __brev(v) >> 21;
}

__global__ __launch_bounds__(128)
void fa_fft_fwd(const float* __restrict__ timeT,
                float* __restrict__ fre, float* __restrict__ fim) {
  __shared__ float sr[2048], si[2048];
  const int row = blockIdx.x, tid = threadIdx.x;
  const float* src = timeT + (size_t)row * LDIM;
  for (int i = tid; i < 2048; i += 128) {
    const unsigned j = brev11((unsigned)i);
    sr[j] = src[i];
    si[j] = 0.f;
  }
  __syncthreads();
  for (int st = 0; st < 11; ++st) {
    const int mh = 1 << st;
    for (int k = tid; k < 1024; k += 128) {
      const int pos = k & (mh - 1);
      const int i   = ((k >> st) << (st + 1)) + pos;
      const int j   = i + mh;
      const float ang = -3.14159265358979f * (float)pos / (float)mh;
      float s, c; __sincosf(ang, &s, &c);
      const float xr = sr[j], xi = si[j];
      const float tr = c * xr - s * xi;
      const float ti = c * xi + s * xr;
      const float ur = sr[i], ui = si[i];
      sr[i] = ur + tr; si[i] = ui + ti;
      sr[j] = ur - tr; si[j] = ui - ti;
    }
    __syncthreads();
  }
  const float scale = 0.0220970869120796f;  // 1/sqrt(2048)
  float* dr = fre + (size_t)row * LFP;
  float* di = fim + (size_t)row * LFP;
  for (int x = tid; x < LFP; x += 128) {
    const bool ok = (x <= 1024);
    dr[x] = ok ? sr[x] * scale : 0.f;
    di[x] = ok ? si[x] * scale : 0.f;
  }
}

// ---------------------------------------------------------------------------
// 3a) Pack Q/K: freq f32 [bh][e][x] -> f16 [bh][x][e]  (e-contiguous rows,
//     zero-filled for x >= 1025). This is the A/B fragment-friendly layout.
// ---------------------------------------------------------------------------
__global__ __launch_bounds__(256)
void fa_pack_qk(const float* __restrict__ fr, const float* __restrict__ fi,
                _Float16* __restrict__ hr, _Float16* __restrict__ hi) {
  __shared__ float tr[32][33];
  __shared__ float ti[32][33];
  const int bh = blockIdx.z;
  const int x0 = blockIdx.x * 32, e0 = blockIdx.y * 32;
  const int tx = threadIdx.x, ty = threadIdx.y;
#pragma unroll
  for (int k = 0; k < 4; ++k) {
    const int e = e0 + ty + 8 * k;
    tr[ty + 8 * k][tx] = fr[((size_t)bh * EDIM + e) * LFP + x0 + tx];
    ti[ty + 8 * k][tx] = fi[((size_t)bh * EDIM + e) * LFP + x0 + tx];
  }
  __syncthreads();
#pragma unroll
  for (int k = 0; k < 4; ++k) {
    const int x = x0 + ty + 8 * k;
    const bool ok = (x < LF);
    const size_t o = ((size_t)bh * LFP + x) * EDIM + e0 + tx;
    hr[o] = ok ? (_Float16)tr[tx][ty + 8 * k] : (_Float16)0.f;
    hi[o] = ok ? (_Float16)ti[tx][ty + 8 * k] : (_Float16)0.f;
  }
}

// 3b) Pack V: straight f32 -> f16 copy, [bh][e][y] (y-contiguous), zero pads.
__global__ __launch_bounds__(256)
void fa_pack_v(const float* __restrict__ fr, const float* __restrict__ fi,
               _Float16* __restrict__ hr, _Float16* __restrict__ hi) {
  const size_t n = (size_t)BH * EDIM * LFP;
  const size_t i = (size_t)blockIdx.x * 256 + threadIdx.x;
  if (i >= n) return;
  const bool ok = (i % LFP) < LF;
  hr[i] = ok ? (_Float16)fr[i] : (_Float16)0.f;
  hi[i] = ok ? (_Float16)fi[i] : (_Float16)0.f;
}

// ---------------------------------------------------------------------------
// 4) Fused frequency-domain attention (flash style), one wave per 16 x-rows.
//    Score tile computed transposed (M=y, N=x): softmax needs only in-lane
//    reductions + one shfl_xor(16); exp-weights are one xor-16 exchange away
//    from the B-fragment layout of the P*V WMMA.  f16 WMMA has no A/B NEG, so
//    Re/Im are formed from 4 independent accumulators (QrKr,QiKi,QiKr,QrKi).
//    Softmax runs in log2 domain (v_exp_f32 is base-2 native); |S| uses the
//    raw v_sqrt_f32 (scores are O(1), never subnormal; feeds a normalized
//    softmax, so the IEEE denormal-rescue sequence is dead weight).
// ---------------------------------------------------------------------------
#define SCORE_SCALE 0.1803368801111204f  // log2(e) / sqrt(E) = 1.44269504/8

__global__ __launch_bounds__(32)
void fa_attention(const _Float16* __restrict__ Qre, const _Float16* __restrict__ Qim,
                  const _Float16* __restrict__ Kre, const _Float16* __restrict__ Kim,
                  const _Float16* __restrict__ Vre, const _Float16* __restrict__ Vim,
                  float* __restrict__ Ore, float* __restrict__ Oim) {
  const int lane = threadIdx.x;
  const int l16  = lane & 15;
  const int half = lane >> 4;
  const int bh   = blockIdx.y;
  const int xBase = blockIdx.x * 16;
  const int x = xBase + l16;
  const float NEG_INF = -__builtin_inff();

  // Q B-fragments (K=e, N=x), loaded once: lane takes e = chunk*32 + half*8 + [0,8) and chunk*32 + 16 + half*8 + [0,8)
  const _Float16* qr = Qre + ((size_t)bh * LFP + x) * EDIM;
  const _Float16* qi = Qim + ((size_t)bh * LFP + x) * EDIM;
  const v16h bqr0 = combine8(qr + half * 8,      qr + 16 + half * 8);
  const v16h bqr1 = combine8(qr + 32 + half * 8, qr + 48 + half * 8);
  const v16h bqi0 = combine8(qi + half * 8,      qi + 16 + half * 8);
  const v16h bqi1 = combine8(qi + 32 + half * 8, qi + 48 + half * 8);

  const _Float16* krBase = Kre + (size_t)bh * LFP * EDIM;
  const _Float16* kiBase = Kim + (size_t)bh * LFP * EDIM;
  const _Float16* vrBase = Vre + (size_t)bh * EDIM * LFP;
  const _Float16* viBase = Vim + (size_t)bh * EDIM * LFP;

  v8f o_re[4] = {};      // O[e,x] accumulators, 4 e-tiles, real
  v8f o_im[4] = {};      // imag
  float mrun = NEG_INF;  // running max, log2 domain
  float ssum = 0.f;

  for (int s = 0; s < NSTEP; ++s) {
    const int yB = s * 32;
    float p0[8], p1[8];

    // ---- two 16x16 S^T tiles (y-tiles t=0,1) ----
#pragma unroll
    for (int t = 0; t < 2; ++t) {
      const int yLoad = yB + t * 16 + l16;                 // A-frag row M=y
      const _Float16* kr = krBase + (size_t)yLoad * EDIM;
      const _Float16* ki = kiBase + (size_t)yLoad * EDIM;
      const v16h akr0 = combine8(kr + half * 8,      kr + 16 + half * 8);
      const v16h akr1 = combine8(kr + 32 + half * 8, kr + 48 + half * 8);
      const v16h aki0 = combine8(ki + half * 8,      ki + 16 + half * 8);
      const v16h aki1 = combine8(ki + 32 + half * 8, ki + 48 + half * 8);

      v8f cu = {};  cu = wmma_f16(akr0, bqr0, cu);  cu = wmma_f16(akr1, bqr1, cu); // Σ Kr·Qr
      v8f cv = {};  cv = wmma_f16(aki0, bqi0, cv);  cv = wmma_f16(aki1, bqi1, cv); // Σ Ki·Qi
      v8f cw = {};  cw = wmma_f16(akr0, bqi0, cw);  cw = wmma_f16(akr1, bqi1, cw); // Σ Kr·Qi
      v8f cz = {};  cz = wmma_f16(aki0, bqr0, cz);  cz = wmma_f16(aki1, bqr1, cz); // Σ Ki·Qr

      float* pt = t ? p1 : p0;
#pragma unroll
      for (int r = 0; r < 8; ++r) {
        const float re = cu[r] + cv[r];                    // Re(S)
        const float im = cw[r] - cz[r];                    // Im(S)
        const int   yy = yB + t * 16 + r + 8 * half;       // element's y (M=r+8*half)
        // log2-domain score: log2e * |S| / sqrt(E); raw v_sqrt_f32
        const float sc = SCORE_SCALE * __builtin_amdgcn_sqrtf(re * re + im * im);
        pt[r] = (yy < LF) ? sc : NEG_INF;
      }
    }

    // ---- online softmax (log2 domain) over this step's 32 y-values ----
    float mloc = p0[0];
#pragma unroll
    for (int r = 1; r < 8; ++r) mloc = fmaxf(mloc, p0[r]);
#pragma unroll
    for (int r = 0; r < 8; ++r) mloc = fmaxf(mloc, p1[r]);
    mloc = fmaxf(mloc, __shfl_xor(mloc, 16, 32));          // join half-wave partner
    const float mnew  = fmaxf(mrun, mloc);
    const float alpha = __builtin_amdgcn_exp2f(mrun - mnew);
    float lsum = 0.f;
#pragma unroll
    for (int r = 0; r < 8; ++r) { p0[r] = __builtin_amdgcn_exp2f(p0[r] - mnew); lsum += p0[r]; }
#pragma unroll
    for (int r = 0; r < 8; ++r) { p1[r] = __builtin_amdgcn_exp2f(p1[r] - mnew); lsum += p1[r]; }
    lsum += __shfl_xor(lsum, 16, 32);
    ssum = ssum * alpha + lsum;
    mrun = mnew;
#pragma unroll
    for (int t = 0; t < 4; ++t)
#pragma unroll
      for (int r = 0; r < 8; ++r) { o_re[t][r] *= alpha; o_im[t][r] *= alpha; }

    // ---- build P^T B-fragment (K=y in [0,32), N=x) via xor-16 exchange ----
    v16h pb;
#pragma unroll
    for (int r = 0; r < 8; ++r) {
      const float x0 = __shfl_xor(p0[r], 16, 32);
      const float x1 = __shfl_xor(p1[r], 16, 32);
      pb[r]     = (_Float16)(half ? x1    : p0[r]);
      pb[r + 8] = (_Float16)(half ? p1[r] : x0);
    }

    // ---- O += V * P^T  (A = V frag: M=e, K=y) ----
#pragma unroll
    for (int t = 0; t < 4; ++t) {
      const int e = t * 16 + l16;
      const _Float16* vr = vrBase + (size_t)e * LFP + yB;
      const _Float16* vi = viBase + (size_t)e * LFP + yB;
      const v16h avr = combine8(vr + half * 16, vr + half * 16 + 8);
      const v16h avi = combine8(vi + half * 16, vi + half * 16 + 8);
      o_re[t] = wmma_f16(avr, pb, o_re[t]);
      o_im[t] = wmma_f16(avi, pb, o_im[t]);
    }

    // prefetch next step's K rows into cache (global_prefetch_b8)
    if (s + 1 < NSTEP) {
      __builtin_prefetch(krBase + (size_t)(yB + 32 + l16) * EDIM, 0, 1);
      __builtin_prefetch(kiBase + (size_t)(yB + 32 + l16) * EDIM, 0, 1);
    }
  }

  const float inv = __builtin_amdgcn_rcpf(ssum);
  if (x < LF) {
#pragma unroll
    for (int t = 0; t < 4; ++t)
#pragma unroll
      for (int r = 0; r < 8; ++r) {
        const int e = t * 16 + r + 8 * half;
        const size_t o = ((size_t)bh * EDIM + e) * LFP + x;
        Ore[o] = o_re[t][r] * inv;
        Oim[o] = o_im[t][r] * inv;
      }
  }
}

// ---------------------------------------------------------------------------
// 5) Hermitian inverse FFT: bins 0..1024 -> real 2048 samples, ortho scale.
// ---------------------------------------------------------------------------
__global__ __launch_bounds__(128)
void fa_fft_inv(const float* __restrict__ fre, const float* __restrict__ fim,
                float* __restrict__ timeT) {
  __shared__ float sr[2048], si[2048];
  const int row = blockIdx.x, tid = threadIdx.x;
  const float* pr = fre + (size_t)row * LFP;
  const float* pi = fim + (size_t)row * LFP;
  for (int i = tid; i < 2048; i += 128) {
    float re, im;
    if (i <= 1024) { re = pr[i];        im =  pi[i]; }
    else           { re = pr[2048 - i]; im = -pi[2048 - i]; }  // conj symmetry
    const unsigned j = brev11((unsigned)i);
    sr[j] = re; si[j] = im;
  }
  __syncthreads();
  for (int st = 0; st < 11; ++st) {
    const int mh = 1 << st;
    for (int k = tid; k < 1024; k += 128) {
      const int pos = k & (mh - 1);
      const int i   = ((k >> st) << (st + 1)) + pos;
      const int j   = i + mh;
      const float ang = 3.14159265358979f * (float)pos / (float)mh; // inverse
      float s, c; __sincosf(ang, &s, &c);
      const float xr = sr[j], xi = si[j];
      const float tr = c * xr - s * xi;
      const float ti = c * xi + s * xr;
      const float ur = sr[i], ui = si[i];
      sr[i] = ur + tr; si[i] = ui + ti;
      sr[j] = ur - tr; si[j] = ui - ti;
    }
    __syncthreads();
  }
  const float scale = 0.0220970869120796f;  // 1/sqrt(2048) (ortho inverse)
  float* dst = timeT + (size_t)row * LDIM;
  for (int i = tid; i < 2048; i += 128) dst[i] = sr[i] * scale;
}

// ---------------------------------------------------------------------------
// 6) Output transpose: timeT [bh][e][l] -> out [b][l][h][e]
// ---------------------------------------------------------------------------
__global__ __launch_bounds__(256)
void fa_transpose_out(const float* __restrict__ src, float* __restrict__ dst) {
  __shared__ float tile[32][33];
  const int bh = blockIdx.z, b = bh >> 3, h = bh & 7;
  const int l0 = blockIdx.x * 32, e0 = blockIdx.y * 32;
  const int tx = threadIdx.x, ty = threadIdx.y;
#pragma unroll
  for (int k = 0; k < 4; ++k) {
    const int e = e0 + ty + 8 * k;
    tile[ty + 8 * k][tx] = src[((size_t)bh * EDIM + e) * LDIM + l0 + tx];  // [e][l]
  }
  __syncthreads();
#pragma unroll
  for (int k = 0; k < 4; ++k) {
    const int l = l0 + ty + 8 * k;
    dst[(((size_t)b * LDIM + l) * HDIM + h) * EDIM + e0 + tx] = tile[tx][ty + 8 * k];
  }
}

// ---------------------------------------------------------------------------
// Launcher
// ---------------------------------------------------------------------------
extern "C" void kernel_launch(void* const* d_in, const int* in_sizes, int n_in,
                              void* d_out, int out_size, void* d_ws, size_t ws_size,
                              hipStream_t stream) {
  (void)in_sizes; (void)n_in; (void)out_size; (void)ws_size;
  const float* q = (const float*)d_in[0];
  const float* k = (const float*)d_in[1];
  const float* v = (const float*)d_in[2];
  float* out = (float*)d_out;

  // Workspace layout (bytes), 256-aligned.
  char* ws = (char*)d_ws;
  size_t off = 0;
  auto take = [&](size_t bytes) -> char* {
    char* p = ws + off;
    off = (off + bytes + 255) & ~(size_t)255;
    return p;
  };
  const size_t timeBytes = (size_t)BH * EDIM * LDIM * sizeof(float);   // 16.8 MB
  const size_t freqBytes = (size_t)BH * EDIM * LFP  * sizeof(float);   //  8.7 MB
  const size_t halfBytes = (size_t)BH * LFP  * EDIM * sizeof(_Float16);//  4.3 MB

  float* timeT = (float*)take(timeBytes);          // reused q,k,v,outT
  float* fRe   = (float*)take(freqBytes);          // reused per tensor; then O_re
  float* fIm   = (float*)take(freqBytes);          //                       O_im
  _Float16* QreH = (_Float16*)take(halfBytes);
  _Float16* QimH = (_Float16*)take(halfBytes);
  _Float16* KreH = (_Float16*)take(halfBytes);
  _Float16* KimH = (_Float16*)take(halfBytes);
  _Float16* VreH = (_Float16*)take(halfBytes);
  _Float16* VimH = (_Float16*)take(halfBytes);

  const dim3 tBlk(32, 8, 1);
  const dim3 tGrdIn(LDIM / 32, EDIM / 32, BH);     // (64, 2, 32)
  const dim3 pGrdQK(LFP / 32, EDIM / 32, BH);      // (33, 2, 32)

  // ---- Q ----
  fa_transpose_in<<<tGrdIn, tBlk, 0, stream>>>(q, timeT);
  fa_fft_fwd<<<BH * EDIM, 128, 0, stream>>>(timeT, fRe, fIm);
  fa_pack_qk<<<pGrdQK, tBlk, 0, stream>>>(fRe, fIm, QreH, QimH);
  // ---- K ----
  fa_transpose_in<<<tGrdIn, tBlk, 0, stream>>>(k, timeT);
  fa_fft_fwd<<<BH * EDIM, 128, 0, stream>>>(timeT, fRe, fIm);
  fa_pack_qk<<<pGrdQK, tBlk, 0, stream>>>(fRe, fIm, KreH, KimH);
  // ---- V ----
  fa_transpose_in<<<tGrdIn, tBlk, 0, stream>>>(v, timeT);
  fa_fft_fwd<<<BH * EDIM, 128, 0, stream>>>(timeT, fRe, fIm);
  {
    const size_t n = (size_t)BH * EDIM * LFP;
    fa_pack_v<<<(unsigned)((n + 255) / 256), 256, 0, stream>>>(fRe, fIm, VreH, VimH);
  }
  // ---- fused frequency-domain attention (writes O spectra into fRe/fIm) ----
  fa_attention<<<dim3(65, BH, 1), 32, 0, stream>>>(QreH, QimH, KreH, KimH,
                                                   VreH, VimH, fRe, fIm);
  // ---- inverse FFT + output transpose ----
  fa_fft_inv<<<BH * EDIM, 128, 0, stream>>>(fRe, fIm, timeT);
  fa_transpose_out<<<tGrdIn, tBlk, 0, stream>>>(timeT, out);
}
